// TokenEmbedding_59313498358359
// MI455X (gfx1250) — matrix operand order repaired and checked by reading, not compiled, not run on hardware
//
#include <hip/hip_runtime.h>
#include <hip/hip_bf16.h>

typedef __attribute__((ext_vector_type(16))) __bf16 v16bf;
typedef __attribute__((ext_vector_type(8)))  float  v8f;

#define H_DIM 768
#define K_ENT 2000
#define M_TOT (64 * 512)
#define LDSW  772   // 768 + 4 pad (floats) to break bank alignment

// ---- fp32 -> bf16 fragment loaders -------------------------------------

__device__ __forceinline__ void cvt8(const float* p, v16bf& d, int at) {
  float4 a = ((const float4*)p)[0];
  float4 b = ((const float4*)p)[1];
  d[at + 0] = (__bf16)a.x; d[at + 1] = (__bf16)a.y;
  d[at + 2] = (__bf16)a.z; d[at + 3] = (__bf16)a.w;
  d[at + 4] = (__bf16)b.x; d[at + 5] = (__bf16)b.y;
  d[at + 6] = (__bf16)b.z; d[at + 7] = (__bf16)b.w;
}

// A-matrix 16x32 bf16 fragment: lane holds 16 contiguous K values [ka..ka+15]
__device__ __forceinline__ v16bf loadA(const float* rowp, int ka) {
  v16bf r;
  cvt8(rowp + ka, r, 0);
  cvt8(rowp + ka + 8, r, 8);
  return r;
}

// B-matrix 32x16 bf16 fragment: lane holds 16 contiguous K values
__device__ __forceinline__ v16bf loadB(const float* p) {
  v16bf r;
  cvt8(p, r, 0);
  cvt8(p + 8, r, 8);
  return r;
}

__device__ __forceinline__ v16bf loadA_guard(const float* rowp, int ka, int kmax) {
  v16bf r;
#pragma unroll
  for (int e = 0; e < 8; ++e) {
    int k1 = ka + e, k2 = ka + 8 + e;
    r[e]     = (k1 < kmax) ? (__bf16)rowp[k1] : (__bf16)0.0f;
    r[e + 8] = (k2 < kmax) ? (__bf16)rowp[k2] : (__bf16)0.0f;
  }
  return r;
}

__device__ __forceinline__ v16bf loadB_guard(const float* rowp, int koff, int kmax) {
  v16bf r;
#pragma unroll
  for (int e = 0; e < 16; ++e) {
    int k = koff + e;
    r[e] = (k < kmax) ? (__bf16)rowp[k] : (__bf16)0.0f;
  }
  return r;
}

// ---- Kernel 1: x = emb @ proj_w^T + proj_b + 5 table gathers ------------
// Block = 256 thr (8 waves). Block tile = 16 rows x 768 cols (full H), so
// emb is streamed from HBM exactly once. Each wave: 6 accumulators (16x16).

__global__ __launch_bounds__(256) void proj_embed_gemm(
    const float* __restrict__ A,     // [M_TOT, K_ENT]
    const float* __restrict__ W,     // [H_DIM, K_ENT]
    const float* __restrict__ pb,    // [H_DIM]
    const int* __restrict__ types, const int* __restrict__ layers,
    const int* __restrict__ ops,   const int* __restrict__ indeg,
    const int* __restrict__ outdeg,
    const float* __restrict__ te, const float* __restrict__ le,
    const float* __restrict__ oe, const float* __restrict__ ie,
    const float* __restrict__ ue,
    float* __restrict__ X)           // [M_TOT, H_DIM]
{
  const int tid   = threadIdx.x;
  const int wv    = tid >> 5;
  const int lane  = tid & 31;
  const int nh    = lane & 15;
  const int mh    = lane >> 4;          // 0/1 lane half
  const int r0    = blockIdx.x * 16;
  const int cbase = wv * 96;            // 8 waves x 96 cols = 768
  const int mrow  = r0 + nh;
  const int h8    = mh * 8;             // unused; A fragments also use h16
  const int h16   = mh * 16;            // B-fragment K half offset

  const float* arow = A + (size_t)mrow * K_ENT;

  v8f acc[6];
#pragma unroll
  for (int j = 0; j < 6; ++j)
#pragma unroll
    for (int e = 0; e < 8; ++e) acc[j][e] = 0.0f;

  // 62 full K-steps of 32 (K = 0..1983)
  for (int kk = 0; kk < 62; ++kk) {
    const int k0 = kk * 32;
    v16bf af = loadA(arow, k0 + h16);
#pragma unroll
    for (int j = 0; j < 6; ++j) {
      const int col = cbase + j * 16 + nh;
      v16bf bf = loadB(W + (size_t)col * K_ENT + k0 + h16);
      acc[j] = __builtin_amdgcn_wmma_f32_16x16x32_bf16(
          false, af, false, bf, (short)0, acc[j], false, false);
    }
  }
  // tail K-step: K = 1984..1999 valid, rest zero-padded
  {
    const int k0 = 1984;
    v16bf af = loadA_guard(arow, k0 + h16, K_ENT);
#pragma unroll
    for (int j = 0; j < 6; ++j) {
      const int col = cbase + j * 16 + nh;
      v16bf bf = loadB_guard(W + (size_t)col * K_ENT, k0 + h16, K_ENT);
      acc[j] = __builtin_amdgcn_wmma_f32_16x16x32_bf16(
          false, af, false, bf, (short)0, acc[j], false, false);
    }
  }

  // Epilogue: +proj_b + table gathers, store fp32 x
#pragma unroll
  for (int i = 0; i < 8; ++i) {
    const int row = r0 + i + mh * 8;
    const int t  = types[row];
    const int l  = layers[row];
    const int o  = ops[row];
    const int di = indeg[row];
    const int dо = outdeg[row];
#pragma unroll
    for (int j = 0; j < 6; ++j) {
      const int col = cbase + j * 16 + nh;
      float add = pb[col]
                + te[t  * H_DIM + col] + le[l  * H_DIM + col]
                + oe[o  * H_DIM + col] + ie[di * H_DIM + col]
                + ue[dо * H_DIM + col];
      X[(size_t)row * H_DIM + col] = acc[j][i] + add;
    }
  }
}

// ---- Kernel 2: y = x @ neg_w^T; select by negs; LayerNorm ---------------
// Block = 256 thr, 16 rows, in-place on X. x tile staged in LDS (fp32);
// WMMA A-fragments come from LDS (ds_load_b128), B from L2-resident neg_w.

__global__ __launch_bounds__(256) void neg_mlp_layernorm(
    const float* __restrict__ NW,   // [H_DIM, H_DIM]
    const float* __restrict__ nb,   // [H_DIM]
    const int* __restrict__ negs,   // [M_TOT]
    const float* __restrict__ g, const float* __restrict__ b,
    float* X)                        // [M_TOT, H_DIM], read+write
{
  __shared__ float xs[16 * LDSW];
  __shared__ float ps[256], ps2[256];
  __shared__ float mur[16], rsr[16];

  const int tid   = threadIdx.x;
  const int wv    = tid >> 5;
  const int lane  = tid & 31;
  const int nh    = lane & 15;
  const int mh    = lane >> 4;
  const int r0    = blockIdx.x * 16;
  const int cbase = wv * 96;

  // Phase 1: stage x tile into LDS (coalesced)
  for (int r = 0; r < 16; ++r)
    for (int c = tid; c < H_DIM; c += 256)
      xs[r * LDSW + c] = X[(size_t)(r0 + r) * H_DIM + c];
  __syncthreads();

  // Phase 2: y = x @ neg_w^T, K = 768 = 24 steps of 32
  v8f acc[6];
#pragma unroll
  for (int j = 0; j < 6; ++j)
#pragma unroll
    for (int e = 0; e < 8; ++e) acc[j][e] = 0.0f;

  const float* xrow = &xs[nh * LDSW];
  for (int kk = 0; kk < 24; ++kk) {
    const int k0 = kk * 32;
    v16bf af = loadA(xrow, k0 + mh * 16);
#pragma unroll
    for (int j = 0; j < 6; ++j) {
      const int col = cbase + j * 16 + nh;
      v16bf bf = loadB(NW + (size_t)col * H_DIM + k0 + mh * 16);
      acc[j] = __builtin_amdgcn_wmma_f32_16x16x32_bf16(
          false, af, false, bf, (short)0, acc[j], false, false);
    }
  }
  __syncthreads();   // all waves done reading xs

  // Phase 3: select (negs==1 ? y+nb : x) back into LDS
#pragma unroll
  for (int i = 0; i < 8; ++i) {
    const int row = i + mh * 8;
    const int nf  = negs[r0 + row];
#pragma unroll
    for (int j = 0; j < 6; ++j) {
      const int col = cbase + j * 16 + nh;
      float yv = acc[j][i] + nb[col];
      float xv = xs[row * LDSW + col];
      xs[row * LDSW + col] = (nf == 1) ? yv : xv;
    }
  }
  __syncthreads();

  // Phase 4: per-row mean/var (16 threads per row, then final reduce)
  {
    const int sub = tid & 15, rl = tid >> 4;
    float s = 0.0f, s2 = 0.0f;
    for (int c = sub; c < H_DIM; c += 16) {
      float v = xs[rl * LDSW + c];
      s += v; s2 += v * v;
    }
    ps[tid] = s; ps2[tid] = s2;
  }
  __syncthreads();
  if (tid < 16) {
    float S = 0.0f, S2 = 0.0f;
#pragma unroll
    for (int e = 0; e < 16; ++e) { S += ps[tid * 16 + e]; S2 += ps2[tid * 16 + e]; }
    const float inv = 1.0f / (float)H_DIM;
    float mu  = S * inv;
    float var = S2 * inv - mu * mu;
    mur[tid] = mu;
    rsr[tid] = rsqrtf(var + 1e-12f);
  }
  __syncthreads();

  // Phase 5: normalize + affine, coalesced store
  for (int r = 0; r < 16; ++r) {
    const float mu = mur[r], rs = rsr[r];
    for (int c = tid; c < H_DIM; c += 256) {
      float v = xs[r * LDSW + c];
      X[(size_t)(r0 + r) * H_DIM + c] = (v - mu) * rs * g[c] + b[c];
    }
  }
}

// ---- Launch -------------------------------------------------------------

extern "C" void kernel_launch(void* const* d_in, const int* in_sizes, int n_in,
                              void* d_out, int out_size, void* d_ws, size_t ws_size,
                              hipStream_t stream) {
  const float* emb    = (const float*)d_in[0];
  const int*   types  = (const int*)d_in[1];
  const int*   layers = (const int*)d_in[2];
  const int*   ops    = (const int*)d_in[3];
  const int*   indeg  = (const int*)d_in[4];
  const int*   outdeg = (const int*)d_in[5];
  const int*   negs   = (const int*)d_in[6];
  const float* te     = (const float*)d_in[7];
  const float* le     = (const float*)d_in[8];
  const float* oe     = (const float*)d_in[9];
  const float* ie     = (const float*)d_in[10];
  const float* ue     = (const float*)d_in[11];
  const float* proj_w = (const float*)d_in[12];
  const float* proj_b = (const float*)d_in[13];
  const float* neg_w  = (const float*)d_in[14];
  const float* neg_b  = (const float*)d_in[15];
  const float* ln_g   = (const float*)d_in[16];
  const float* ln_b   = (const float*)d_in[17];
  float* X = (float*)d_out;

  proj_embed_gemm<<<M_TOT / 16, 256, 0, stream>>>(
      emb, proj_w, proj_b, types, layers, ops, indeg, outdeg,
      te, le, oe, ie, ue, X);

  neg_mlp_layernorm<<<M_TOT / 16, 256, 0, stream>>>(
      neg_w, neg_b, negs, ln_g, ln_b, X);
}
